// GCNPolicy_11433202942390
// MI455X (gfx1250) — matrix operand order repaired and checked by reading, not compiled, not run on hardware
//
#include <hip/hip_runtime.h>

#define Nn 131072
#define Ee 1600000
#define Bb 64
#define Hh 128
#define Ll 3
#define Kk 5
#define MTt 20

typedef __attribute__((ext_vector_type(16))) _Float16 v16h;
typedef __attribute__((ext_vector_type(8)))  float    v8f;
typedef int vi4 __attribute__((vector_size(16)));   // matches builtin's V4i param

// fragment: 16 halves = 8 dwords = 2 x 16-byte quads (matches the piecewise-
// contiguous ISA layouts so loads become ds_load_b128 / global_load_b128)
union HFrag { v16h v; unsigned u[8]; uint4 q[2]; };

// ---------------- async global->LDS staging (CDNA5 GLOBAL_LOAD_ASYNC_TO_LDS) ----
#if __has_builtin(__builtin_amdgcn_global_load_async_to_lds_b128)
#define HAVE_ASYNC_COPY 1
#endif

__device__ __forceinline__ void copy16_g2l(const void* gsrc, void* ldst) {
#if HAVE_ASYNC_COPY
    __builtin_amdgcn_global_load_async_to_lds_b128(
        (__attribute__((address_space(1))) vi4*)gsrc,
        (__attribute__((address_space(3))) vi4*)ldst, 0, 0);
#else
    *(uint4*)ldst = *(const uint4*)gsrc;
#endif
}

__device__ __forceinline__ void copy16_wait() {
#if HAVE_ASYNC_COPY
#if __has_builtin(__builtin_amdgcn_s_wait_asynccnt)
    __builtin_amdgcn_s_wait_asynccnt(0);
#else
    asm volatile("s_wait_asynccnt 0x0" ::: "memory");
#endif
#endif
}

// ---------------------------------------------------------------- utilities
__global__ void fill_f32(float* p, float v, int n) {
    int i = blockIdx.x * 256 + threadIdx.x;
    if (i < n) p[i] = v;
}

__global__ void f32_to_f16_k(_Float16* d, const float* s, int n) {
    int i = blockIdx.x * 256 + threadIdx.x;
    if (i < n) d[i] = (_Float16)s[i];
}

// deg[dst] += 1 over edges (self-loop handled by init deg=1)
__global__ void deg_edges_k(const int* __restrict__ dst, float* __restrict__ deg, int E) {
    int e = blockIdx.x * 256 + threadIdx.x;
    if (e < E) unsafeAtomicAdd(&deg[dst[e]], 1.0f);
}

__global__ void rsqrt_ip_k(float* p, int n) {
    int i = blockIdx.x * 256 + threadIdx.x;
    if (i < n) p[i] = p[i] > 0.0f ? rsqrtf(p[i]) : 0.0f;
}

// x [N,16] f32 -> x16 [N,32] f16 zero padded
__global__ void pad_x16_k(_Float16* __restrict__ xp, const float* __restrict__ x) {
    int i = blockIdx.x * 256 + threadIdx.x;   // N*32 threads
    int node = i >> 5, c = i & 31;
    xp[i] = (c < 16) ? (_Float16)x[node * 16 + c] : (_Float16)0.0f;
}

// dst[b][n][c] = (f16) src[b][c][n], with zero pad c>=C (dst has Cpad cols)
__global__ void transpose_pad_f16_k(_Float16* __restrict__ dst, const float* __restrict__ src,
                                    int C, int Cpad, int Ncol, int total) {
    int i = blockIdx.x * 256 + threadIdx.x;
    if (i >= total) return;
    int per = Ncol * Cpad;
    int bidx = i / per, r = i % per;
    int n = r / Cpad, c = r % Cpad;
    dst[i] = (c < C) ? (_Float16)src[(size_t)bidx * C * Ncol + (size_t)c * Ncol + n]
                     : (_Float16)0.0f;
}

// ------------------------------------------------- WMMA GEMM: C = act(A @ B + bias)
// A: [N][KDIM] f16 row-major; Bt: [128][KDIM] f16 (weights pre-transposed)
// One block = 16 rows x 128 cols; 8 waves, each owns a 16x16 tile over all of K.
template <int KDIM, bool RELU, bool BIAS>
__global__ void __launch_bounds__(256)
wmma_gemm_k(const _Float16* __restrict__ A, const _Float16* __restrict__ Bt,
            const float* __restrict__ bias, float* __restrict__ C) {
    __shared__ __align__(16) unsigned atile[16 * KDIM / 2];
    const int tid  = threadIdx.x;
    const int wave = tid >> 5, lane = tid & 31;
    const int rowBase = blockIdx.x * 16;

    // stage A tile (16 x KDIM halves) to LDS via async 16B transfers
    constexpr int Q = KDIM / 8;                 // uint4s per row
    const uint4* Aq = (const uint4*)A;
    uint4* atq = (uint4*)atile;
    for (int i = tid; i < 16 * Q; i += 256) {
        int r = i / Q, c = i % Q;
        copy16_g2l(Aq + (size_t)(rowBase + r) * Q + c, atq + i);
    }
    copy16_wait();
    __syncthreads();

    const int m   = lane & 15;       // A row within tile / B col within tile
    const int sel = lane >> 4;
    const int colBase = wave * 16;
    const uint4* Bq = (const uint4*)Bt;

    HFrag a, b;
    v8f acc = {};
#pragma unroll
    for (int step = 0; step < KDIM / 32; ++step) {
        const int kk0 = step * 32;
        // A 16x32 f16 ISA layout: per lane j=0..3 -> halves [sel*8, sel*8+7],
        // j=4..7 -> +16 halves; both 16B-aligned quads.
        int abase = (m * KDIM + kk0 + sel * 8) >> 3;  // in uint4 units
        a.q[0] = atq[abase];
        a.q[1] = atq[abase + 2];
        // B KxN f16 layout matching A: halves [sel*8, sel*8+7] and +16 halves.
        size_t bbase = ((size_t)(colBase + m) * KDIM + kk0 + sel * 8) >> 3;
        b.q[0] = Bq[bbase];
        b.q[1] = Bq[bbase + 2];
        acc = __builtin_amdgcn_wmma_f32_16x16x32_f16(false, a.v, false, b.v,
                                                     (short)0, acc, false, false);
    }
#pragma unroll
    for (int r = 0; r < 8; ++r) {
        int mm = r + 8 * sel;                 // D layout: VGPR r -> M=r (+8 for hi lanes)
        float v = acc[r];
        if (BIAS) v += bias[colBase + m];
        if (RELU) v = fmaxf(v, 0.0f);
        C[(size_t)(rowBase + mm) * Hh + colBase + m] = v;
    }
}

// ------------------------------------------- fused policy head: logits = relu(A@W1+b1)@W2 + b2
__global__ void __launch_bounds__(256)
policy_head_k(const _Float16* __restrict__ comb,   // [N][256]
              const _Float16* __restrict__ W1t,    // [K][128][256]
              const float* __restrict__ b1,        // [K][128]
              const float* __restrict__ W2,        // [K][128]
              const float* __restrict__ b2,        // [K]
              float* __restrict__ out) {           // [K][N]
    constexpr int KDIM = 2 * Hh;                   // 256
    __shared__ __align__(16) unsigned atile[16 * KDIM / 2];
    __shared__ float red[16];
    const int tid  = threadIdx.x;
    const int wave = tid >> 5, lane = tid & 31;
    const int rowBase = blockIdx.x * 16;
    const int head = blockIdx.y;

    constexpr int Q = KDIM / 8;
    const uint4* Aq = (const uint4*)comb;
    uint4* atq = (uint4*)atile;
    for (int i = tid; i < 16 * Q; i += 256) {
        int r = i / Q, c = i % Q;
        copy16_g2l(Aq + (size_t)(rowBase + r) * Q + c, atq + i);
    }
    if (tid < 16) red[tid] = 0.0f;
    copy16_wait();
    __syncthreads();

    const int m   = lane & 15;
    const int sel = lane >> 4;
    const int colBase = wave * 16;
    const uint4* Bq = (const uint4*)(W1t + (size_t)head * Hh * KDIM);

    HFrag a, b;
    v8f acc = {};
#pragma unroll
    for (int step = 0; step < KDIM / 32; ++step) {
        const int kk0 = step * 32;
        int abase = (m * KDIM + kk0 + sel * 8) >> 3;
        a.q[0] = atq[abase];
        a.q[1] = atq[abase + 2];
        size_t bbase = ((size_t)(colBase + m) * KDIM + kk0 + sel * 8) >> 3;
        b.q[0] = Bq[bbase];
        b.q[1] = Bq[bbase + 2];
        acc = __builtin_amdgcn_wmma_f32_16x16x32_f16(false, a.v, false, b.v,
                                                     (short)0, acc, false, false);
    }
    const float bb  = b1[head * Hh + colBase + m];
    const float w2v = W2[head * Hh + colBase + m];
#pragma unroll
    for (int r = 0; r < 8; ++r) {
        int mm = r + 8 * sel;
        float v = fmaxf(acc[r] + bb, 0.0f);
        atomicAdd(&red[mm], v * w2v);          // ds_add_f32
    }
    __syncthreads();
    if (tid < 16)
        out[(size_t)head * Nn + rowBase + tid] = red[tid] + b2[head];
}

// ---------------------------------------------------------------- graph ops
__global__ void scatter_edges_k(const float* __restrict__ m, const int* __restrict__ src,
                                const int* __restrict__ dst, const float* __restrict__ dinv,
                                float* __restrict__ agg, int E) {
    int t = blockIdx.x * 256 + threadIdx.x;    // E*32 threads, 4 feats each
    int e = t >> 5;
    if (e >= E) return;
    int g = (t & 31) << 2;
    int s = src[e], d = dst[e];
    float nm = dinv[s] * dinv[d];
    const float4 mv = *(const float4*)(m + (size_t)s * Hh + g);
    float* ap = agg + (size_t)d * Hh + g;
    unsafeAtomicAdd(ap + 0, mv.x * nm);
    unsafeAtomicAdd(ap + 1, mv.y * nm);
    unsafeAtomicAdd(ap + 2, mv.z * nm);
    unsafeAtomicAdd(ap + 3, mv.w * nm);
}

// h = relu(LN(agg + m*dinv^2 + conv_b)) + h   (one wave per node)
__global__ void post_ln_k(const float* __restrict__ agg, const float* __restrict__ m,
                          const float* __restrict__ dinv, const float* __restrict__ cb,
                          const float* __restrict__ lg, const float* __restrict__ lb,
                          float* __restrict__ h) {
    int wave = threadIdx.x >> 5, lane = threadIdx.x & 31;
    int node = blockIdx.x * 8 + wave;
    if (node >= Nn) return;
    size_t base = (size_t)node * Hh;
    float sl = dinv[node]; sl *= sl;
    int f0 = lane * 4;
    float4 av = *(const float4*)(agg + base + f0);
    float4 mv = *(const float4*)(m + base + f0);
    float v0 = av.x + mv.x * sl + cb[f0 + 0];
    float v1 = av.y + mv.y * sl + cb[f0 + 1];
    float v2 = av.z + mv.z * sl + cb[f0 + 2];
    float v3 = av.w + mv.w * sl + cb[f0 + 3];
    float s = v0 + v1 + v2 + v3;
    for (int k = 16; k; k >>= 1) s += __shfl_xor(s, k, 32);
    float mean = s * (1.0f / Hh);
    float d0 = v0 - mean, d1 = v1 - mean, d2 = v2 - mean, d3 = v3 - mean;
    float q = d0 * d0 + d1 * d1 + d2 * d2 + d3 * d3;
    for (int k = 16; k; k >>= 1) q += __shfl_xor(q, k, 32);
    float rstd = rsqrtf(q * (1.0f / Hh) + 1e-5f);
    float4 res = *(const float4*)(h + base + f0);
    float4 o;
    o.x = fmaxf(d0 * rstd * lg[f0 + 0] + lb[f0 + 0], 0.0f) + res.x;
    o.y = fmaxf(d1 * rstd * lg[f0 + 1] + lb[f0 + 1], 0.0f) + res.y;
    o.z = fmaxf(d2 * rstd * lg[f0 + 2] + lb[f0 + 2], 0.0f) + res.z;
    o.w = fmaxf(d3 * rstd * lg[f0 + 3] + lb[f0 + 3], 0.0f) + res.w;
    *(float4*)(h + base + f0) = o;
}

__global__ void pool_cnt_k(const int* __restrict__ batch, float* __restrict__ counts) {
    int i = blockIdx.x * 256 + threadIdx.x;
    if (i < Nn) unsafeAtomicAdd(&counts[batch[i]], 1.0f);
}

__global__ void pool_sum_k(const float* __restrict__ h, const int* __restrict__ batch,
                           float* __restrict__ gs) {
    int t = blockIdx.x * 256 + threadIdx.x;    // N*32 threads
    int node = t >> 5;
    if (node >= Nn) return;
    int g = (t & 31) << 2;
    int b = batch[node];
    const float4 hv = *(const float4*)(h + (size_t)node * Hh + g);
    float* p = gs + b * Hh + g;
    unsafeAtomicAdd(p + 0, hv.x);
    unsafeAtomicAdd(p + 1, hv.y);
    unsafeAtomicAdd(p + 2, hv.z);
    unsafeAtomicAdd(p + 3, hv.w);
}

__global__ void pool_div_k(const float* __restrict__ gs, const float* __restrict__ counts,
                           float* __restrict__ ge) {
    int i = blockIdx.x * 256 + threadIdx.x;
    if (i >= Bb * Hh) return;
    ge[i] = gs[i] / fmaxf(counts[i / Hh], 1.0f);
}

__global__ void glob_emb_k(const float* __restrict__ gf, const float* __restrict__ W,
                           const float* __restrict__ bv, float* __restrict__ ge,
                           _Float16* __restrict__ ge16) {
    int i = blockIdx.x * 256 + threadIdx.x;
    if (i >= Bb * Hh) return;
    int b = i / Hh, f = i % Hh;
    float acc = bv[f];
    for (int c = 0; c < 32; ++c) acc += gf[b * 32 + c] * W[c * Hh + f];
    acc = fmaxf(acc, 0.0f);
    ge[i] = acc; ge16[i] = (_Float16)acc;
}

__global__ void build_comb_k(const _Float16* __restrict__ h16, const _Float16* __restrict__ ge16,
                             const int* __restrict__ batch, _Float16* __restrict__ comb) {
    int i = blockIdx.x * 256 + threadIdx.x;    // N*256 threads
    if (i >= Nn * 256) return;
    int node = i >> 8, f = i & 255;
    comb[i] = (f < Hh) ? h16[((size_t)node << 7) + f]
                       : ge16[batch[node] * Hh + (f - Hh)];
}

// ---------------------------------------------------------------- value head
__global__ void value_k(const float* __restrict__ gemb, const float* __restrict__ ge,
                        const float* __restrict__ vng, const float* __restrict__ vnb,
                        const float* __restrict__ v1W, const float* __restrict__ v1b,
                        const float* __restrict__ v2W, const float* __restrict__ v2b,
                        const float* __restrict__ v3W, const float* __restrict__ v3b,
                        float* __restrict__ out) {
    __shared__ float vi[256], v1s[128], v2s[64], wsum[8], shm[2];
    int b = blockIdx.x, tid = threadIdx.x;
    float xv = (tid < Hh) ? gemb[b * Hh + tid] : ge[b * Hh + tid - Hh];
    float s = xv;
    for (int k = 16; k; k >>= 1) s += __shfl_xor(s, k, 32);
    if ((tid & 31) == 0) wsum[tid >> 5] = s;
    __syncthreads();
    if (tid == 0) { float t = 0; for (int i = 0; i < 8; ++i) t += wsum[i]; shm[0] = t * (1.0f / 256.0f); }
    __syncthreads();
    float mean = shm[0];
    float d = xv - mean;
    float q = d * d;
    for (int k = 16; k; k >>= 1) q += __shfl_xor(q, k, 32);
    __syncthreads();
    if ((tid & 31) == 0) wsum[tid >> 5] = q;
    __syncthreads();
    if (tid == 0) { float t = 0; for (int i = 0; i < 8; ++i) t += wsum[i]; shm[1] = rsqrtf(t * (1.0f / 256.0f) + 1e-5f); }
    __syncthreads();
    vi[tid] = d * shm[1] * vng[tid] + vnb[tid];
    __syncthreads();
    if (tid < 128) {
        float acc = v1b[tid];
        for (int c = 0; c < 256; ++c) acc += vi[c] * v1W[c * 128 + tid];
        v1s[tid] = fmaxf(acc, 0.0f);
    }
    __syncthreads();
    if (tid < 64) {
        float acc = v2b[tid];
        for (int c = 0; c < 128; ++c) acc += v1s[c] * v2W[c * 64 + tid];
        v2s[tid] = fmaxf(acc, 0.0f);
    }
    __syncthreads();
    if (tid == 0) {
        float acc = v3b[0];
        for (int c = 0; c < 64; ++c) acc += v2s[c] * v3W[c];
        out[b] = acc;
    }
}

// ---------------------------------------------------------------- troops head
__global__ void troops_k(const float* __restrict__ ge, const float* __restrict__ W1,
                         const float* __restrict__ b1, const float* __restrict__ W2,
                         const float* __restrict__ b2, float* __restrict__ out) {
    __shared__ float th[128];
    int k = blockIdx.x / Bb, b = blockIdx.x % Bb, tid = threadIdx.x;
    float acc = b1[k * Hh + tid];
    for (int c = 0; c < Hh; ++c) acc += ge[b * Hh + c] * W1[(size_t)k * Hh * Hh + c * Hh + tid];
    th[tid] = fmaxf(acc, 0.0f);
    __syncthreads();
    if (tid < MTt) {
        float o = b2[k * MTt + tid];
        for (int d = 0; d < Hh; ++d) o += th[d] * W2[(size_t)k * Hh * MTt + d * MTt + tid];
        out[(size_t)(k * Bb + b) * MTt + tid] = o;
    }
}

// ---------------------------------------------------------------- launcher
extern "C" void kernel_launch(void* const* d_in, const int* in_sizes, int n_in,
                              void* d_out, int out_size, void* d_ws, size_t ws_size,
                              hipStream_t stream) {
    const float* x      = (const float*)d_in[0];
    const int*   ei     = (const int*)d_in[1];
    const float* gf     = (const float*)d_in[2];
    const int*   batch  = (const int*)d_in[3];
    const float* in_W   = (const float*)d_in[4];
    const float* in_b   = (const float*)d_in[5];
    const float* conv_W = (const float*)d_in[6];
    const float* conv_b = (const float*)d_in[7];
    const float* ln_g   = (const float*)d_in[8];
    const float* ln_b   = (const float*)d_in[9];
    const float* glob_W = (const float*)d_in[10];
    const float* glob_b = (const float*)d_in[11];
    const float* vn_g   = (const float*)d_in[12];
    const float* vn_b   = (const float*)d_in[13];
    const float* v1_W   = (const float*)d_in[14];
    const float* v1_b   = (const float*)d_in[15];
    const float* v2_W   = (const float*)d_in[16];
    const float* v2_b   = (const float*)d_in[17];
    const float* v3_W   = (const float*)d_in[18];
    const float* v3_b   = (const float*)d_in[19];
    const float* src_W1 = (const float*)d_in[20];
    const float* src_b1 = (const float*)d_in[21];
    const float* src_W2 = (const float*)d_in[22];
    const float* src_b2 = (const float*)d_in[23];
    const float* dst_W1 = (const float*)d_in[24];
    const float* dst_b1 = (const float*)d_in[25];
    const float* dst_W2 = (const float*)d_in[26];
    const float* dst_b2 = (const float*)d_in[27];
    const float* trp_W1 = (const float*)d_in[28];
    const float* trp_b1 = (const float*)d_in[29];
    const float* trp_W2 = (const float*)d_in[30];
    const float* trp_b2 = (const float*)d_in[31];
    float* out = (float*)d_out;

    // workspace carve
    char* base = (char*)d_ws;
    size_t off = 0;
    auto carve = [&](size_t bytes) -> void* {
        void* p = base + off;
        off = (off + bytes + 255) & ~(size_t)255;
        return p;
    };
    float*     deg     = (float*)carve((size_t)Nn * 4);
    float*     h       = (float*)carve((size_t)Nn * Hh * 4);
    float*     mbuf    = (float*)carve((size_t)Nn * Hh * 4);
    float*     agg     = (float*)carve((size_t)Nn * Hh * 4);
    _Float16*  h16     = (_Float16*)carve((size_t)Nn * Hh * 2);
    _Float16*  x16     = (_Float16*)carve((size_t)Nn * 32 * 2);
    _Float16*  comb16  = (_Float16*)carve((size_t)Nn * 256 * 2);
    float*     gsum    = (float*)carve((size_t)Bb * Hh * 4);
    float*     counts  = (float*)carve((size_t)Bb * 4);
    float*     gemb    = (float*)carve((size_t)Bb * Hh * 4);
    float*     ge      = (float*)carve((size_t)Bb * Hh * 4);
    _Float16*  ge16    = (_Float16*)carve((size_t)Bb * Hh * 2);
    _Float16*  inWt    = (_Float16*)carve((size_t)128 * 32 * 2);
    _Float16*  convWt  = (_Float16*)carve((size_t)3 * 128 * 128 * 2);
    _Float16*  srcW1t  = (_Float16*)carve((size_t)5 * 128 * 256 * 2);
    _Float16*  dstW1t  = (_Float16*)carve((size_t)5 * 128 * 256 * 2);

    auto blocks = [](long long n) { return (unsigned)((n + 255) / 256); };

    // GCN degree norm
    fill_f32<<<blocks(Nn), 256, 0, stream>>>(deg, 1.0f, Nn);
    deg_edges_k<<<blocks(Ee), 256, 0, stream>>>(ei + Ee, deg, Ee);
    rsqrt_ip_k<<<blocks(Nn), 256, 0, stream>>>(deg, Nn);

    // weight prep (f16, transposed)
    pad_x16_k<<<blocks((long long)Nn * 32), 256, 0, stream>>>(x16, x);
    transpose_pad_f16_k<<<blocks(128 * 32), 256, 0, stream>>>(inWt, in_W, 16, 32, 128, 128 * 32);
    transpose_pad_f16_k<<<blocks(3 * 128 * 128), 256, 0, stream>>>(convWt, conv_W, 128, 128, 128, 3 * 128 * 128);
    transpose_pad_f16_k<<<blocks(5 * 128 * 256), 256, 0, stream>>>(srcW1t, src_W1, 256, 256, 128, 5 * 128 * 256);
    transpose_pad_f16_k<<<blocks(5 * 128 * 256), 256, 0, stream>>>(dstW1t, dst_W1, 256, 256, 128, 5 * 128 * 256);

    // input linear: h = relu(x @ in_W + b)   [WMMA, K=32]
    wmma_gemm_k<32, true, true><<<Nn / 16, 256, 0, stream>>>(x16, inWt, in_b, h);
    f32_to_f16_k<<<blocks((long long)Nn * Hh), 256, 0, stream>>>(h16, h, Nn * Hh);

    for (int l = 0; l < Ll; ++l) {
        wmma_gemm_k<128, false, false><<<Nn / 16, 256, 0, stream>>>(
            h16, convWt + (size_t)l * 128 * 128, nullptr, mbuf);
        fill_f32<<<blocks((long long)Nn * Hh), 256, 0, stream>>>(agg, 0.0f, Nn * Hh);
        scatter_edges_k<<<blocks((long long)Ee * 32), 256, 0, stream>>>(mbuf, ei, ei + Ee, deg, agg, Ee);
        post_ln_k<<<Nn / 8, 256, 0, stream>>>(agg, mbuf, deg, conv_b + l * Hh,
                                              ln_g + l * Hh, ln_b + l * Hh, h);
        f32_to_f16_k<<<blocks((long long)Nn * Hh), 256, 0, stream>>>(h16, h, Nn * Hh);
    }

    // mean pool + global emb
    fill_f32<<<blocks(Bb * Hh), 256, 0, stream>>>(gsum, 0.0f, Bb * Hh);
    fill_f32<<<1, 256, 0, stream>>>(counts, 0.0f, Bb);
    pool_cnt_k<<<blocks(Nn), 256, 0, stream>>>(batch, counts);
    pool_sum_k<<<blocks((long long)Nn * 32), 256, 0, stream>>>(h, batch, gsum);
    pool_div_k<<<blocks(Bb * Hh), 256, 0, stream>>>(gsum, counts, gemb);
    glob_emb_k<<<blocks(Bb * Hh), 256, 0, stream>>>(gf, glob_W, glob_b, ge, ge16);

    // combined features
    build_comb_k<<<blocks((long long)Nn * 256), 256, 0, stream>>>(h16, ge16, batch, comb16);

    // output offsets
    const size_t srcOff = 0;
    const size_t dstOff = (size_t)Kk * Nn;
    const size_t trpOff = 2 * (size_t)Kk * Nn;
    const size_t valOff = trpOff + (size_t)Kk * Bb * MTt;

    // value + troops (tiny heads)
    value_k<<<Bb, 256, 0, stream>>>(gemb, ge, vn_g, vn_b, v1_W, v1_b, v2_W, v2_b,
                                    v3_W, v3_b, out + valOff);
    troops_k<<<Kk * Bb, 128, 0, stream>>>(ge, trp_W1, trp_b1, trp_W2, trp_b2, out + trpOff);

    // fused policy heads [WMMA, K=256]
    dim3 pgrid(Nn / 16, Kk);
    policy_head_k<<<pgrid, 256, 0, stream>>>(comb16, srcW1t, src_b1, src_W2, src_b2, out + srcOff);
    policy_head_k<<<pgrid, 256, 0, stream>>>(comb16, dstW1t, dst_b1, dst_W2, dst_b2, out + dstOff);
}
